// CrossTrans_58385785422162
// MI455X (gfx1250) — compile-verified, matrix operand order checked by reading, not run on hardware
//
#include <hip/hip_runtime.h>
#include <hip/hip_bf16.h>

// ---------------------------------------------------------------------------
// All-WMMA LeViT attention block for MI455X (gfx1250, wave32).
//
// Central trick: for V_WMMA_F32_16X16X32_F16, the C/D register layout maps
// element (M = v + 8*(lane>=16), N = lane%16) to VGPR slot v of each lane,
// and BOTH the A and B 16-bit operand layouts map
//   slot(lane, i) = element(row = lane%16, k = i + 8*(lane>=16) + (i>=8?8:0)).
// Hence a D tile whose M-axis is the NEXT gemm's contraction axis can be
// stored as that gemm's packed operand with slot i = v, SAME lane:
// one contiguous 16/32-byte store per lane. The whole chain
//   x -> {q,k,v} -> attention -> p(+residual) -> o
// runs on WMMA with f16 operands / f32 accumulation; BN scales fold into the
// packed weights, biases add in epilogues.
// ---------------------------------------------------------------------------

#define BATCH   4
#define CHAN    384
#define NH      6
#define KD      16
#define DHEAD   32
#define NPIX    2304               // 48*48
#define DIMS    96                 // C/4
#define NT      144                // 16-pixel tiles (NPIX/16)
#define XKS     12                 // 32-channel k-steps over 384 input channels
#define OPSZ    512                // one packed operand tile: 32 lanes * 16 halfs

typedef __attribute__((ext_vector_type(16))) _Float16 v16h;
typedef __attribute__((ext_vector_type(8)))  _Float16 v8h;
typedef __attribute__((ext_vector_type(8)))  float    v8f;

// k index held by operand slot (lane,i)  (ISA 7.12.2, 16-bit 16x32 A/B map)
__device__ __forceinline__ int kmap(int i, int lh) {
    return i + 8 * lh + ((i >= 8) ? 8 : 0);
}

#define WMMA(A, B, C) __builtin_amdgcn_wmma_f32_16x16x32_f16( \
        false, (A), false, (B), (short)0, (C), false, false)

// ---------------------------------------------------------------------------
// pack_x : xx (fp32, [b][ci][n]) -> f16 operand tiles xpk[b][nt][ks][512].
// Serves as B operand (K/Q gemms) and A operand (V^T gemm) — same layout.
// ---------------------------------------------------------------------------
__global__ __launch_bounds__(256) void pack_x_kernel(
    const float* __restrict__ xx, _Float16* __restrict__ xpk)
{
    size_t idx = (size_t)blockIdx.x * 256 + threadIdx.x;
    const size_t total = (size_t)BATCH * NT * XKS * OPSZ;
    if (idx >= total) return;
    int i    = idx & 15;
    int lane = (idx >> 4) & 31;
    size_t t = idx >> 9;
    int ks = (int)(t % XKS); t /= XKS;
    int nt = (int)(t % NT);
    int b  = (int)(t / NT);
    int ci = ks * 32 + kmap(i, lane >> 4);
    int n  = nt * 16 + (lane & 15);
    xpk[idx] = (_Float16)xx[((size_t)b * CHAN + ci) * NPIX + n];
}

// ---------------------------------------------------------------------------
// pack_w : generic scale-folded weight packer.
// dst tile (ot, ks): slot(lane,i) = W[ot*16 + lane%16][ks*32 + kmap] * s[oc].
// Works for A tiles (row = out-channel) and B tiles of W^T (row = out-channel).
// ---------------------------------------------------------------------------
__global__ __launch_bounds__(256) void pack_w_kernel(
    const float* __restrict__ W, const float* __restrict__ s,
    _Float16* __restrict__ dst, int OCT, int KS, int CI)
{
    size_t idx = (size_t)blockIdx.x * 256 + threadIdx.x;
    const size_t total = (size_t)OCT * KS * OPSZ;
    if (idx >= total) return;
    int i    = idx & 15;
    int lane = (idx >> 4) & 31;
    size_t t = idx >> 9;
    int ks = (int)(t % KS);
    int ot = (int)(t / KS);
    int oc = ot * 16 + (lane & 15);
    int ci = ks * 32 + kmap(i, lane >> 4);
    dst[idx] = (ci < CI) ? (_Float16)(W[(size_t)oc * CI + ci] * s[oc])
                         : (_Float16)0.0f;
}

// ---------------------------------------------------------------------------
// gemm_qkv : one wave per (b, n-tile, group).
//   g0: K (6 oc-tiles x 12 ksteps) + Q (6 oc-tiles x ksteps 9..11)
//   g1: V^T d-tiles 0..5, g2: V^T d-tiles 6..11   (A = x tile, B = Wv^T tile)
// Epilogues store attention operands directly (slot i = v, same lane).
// ---------------------------------------------------------------------------
__global__ __launch_bounds__(256) void gemm_qkv_kernel(
    const _Float16* __restrict__ xpk,
    const _Float16* __restrict__ wkp, const _Float16* __restrict__ wqp,
    const _Float16* __restrict__ wvp,
    const float* __restrict__ kbias, const float* __restrict__ qbias,
    const float* __restrict__ vbias,
    _Float16* __restrict__ kpk, _Float16* __restrict__ qpk,
    _Float16* __restrict__ vpk)
{
    const int wave = threadIdx.x >> 5, lane = threadIdx.x & 31;
    const int gw = blockIdx.x * 8 + wave;           // 1728 waves
    const int g  = gw % 3;
    const int nt = (gw / 3) % NT;
    const int b  = gw / (3 * NT);
    const int lh = lane >> 4, lanelo = lane & 15;
    const _Float16* xbase = xpk + ((size_t)b * NT + nt) * XKS * OPSZ + lane * 16;

    if (g == 0) {
        v8f accK[6] = {{0}}; v8f accQ[6] = {{0}};
        for (int ks = 0; ks < XKS; ++ks) {
            v16h xop = *(const v16h*)(xbase + (size_t)ks * OPSZ);
            #pragma unroll
            for (int t = 0; t < 6; ++t) {
                v16h wa = *(const v16h*)(wkp + ((size_t)t * XKS + ks) * OPSZ + lane * 16);
                accK[t] = WMMA(wa, xop, accK[t]);
            }
            if (ks >= 9) {                          // Q contracts over ci 288..383
                #pragma unroll
                for (int t = 0; t < 6; ++t) {
                    v16h wa = *(const v16h*)(wqp + ((size_t)t * 3 + (ks - 9)) * OPSZ + lane * 16);
                    accQ[t] = WMMA(wa, xop, accQ[t]);
                }
            }
        }
        #pragma unroll
        for (int t = 0; t < 6; ++t) {               // head h = t (oc tile == KD)
            const float* bp = kbias + t * 16 + 8 * lh;
            v16h e = {};                            // slots 8..15 = K-pad zeros
            #pragma unroll
            for (int v = 0; v < 8; ++v) e[v] = (_Float16)(accK[t][v] + bp[v]);
            *(v16h*)(kpk + (((size_t)(b * NH + t)) * NT + nt) * OPSZ + lane * 16) = e;
        }
        #pragma unroll
        for (int t = 0; t < 6; ++t) {
            const float* bp = qbias + t * 16 + 8 * lh;
            v16h e = {};
            #pragma unroll
            for (int v = 0; v < 8; ++v) e[v] = (_Float16)(accQ[t][v] + bp[v]);
            *(v16h*)(qpk + (((size_t)(b * NH + t)) * NT + nt) * OPSZ + lane * 16) = e;
        }
    } else {
        const int dgbase = (g == 1) ? 0 : 6;
        v8f accV[6] = {{0}};
        for (int ks = 0; ks < XKS; ++ks) {
            v16h xop = *(const v16h*)(xbase + (size_t)ks * OPSZ);
            #pragma unroll
            for (int j = 0; j < 6; ++j) {
                v16h wb = *(const v16h*)(wvp + ((size_t)(dgbase + j) * XKS + ks) * OPSZ + lane * 16);
                accV[j] = WMMA(xop, wb, accV[j]);   // D(M=key, N=d)
            }
        }
        #pragma unroll
        for (int j = 0; j < 6; ++j) {
            const int dg = dgbase + j;
            const int h = dg >> 1, dsel = dg & 1;
            const float bias = vbias[dg * 16 + lanelo];
            v8h e;
            #pragma unroll
            for (int v = 0; v < 8; ++v) e[v] = (_Float16)(accV[j][v] + bias);
            // operand covers 32 keys: n-tile parity selects slots 0..7 / 8..15
            *(v8h*)(vpk + ((((size_t)(b * NH + h)) * (NT / 2) + (nt >> 1)) * 2 + dsel) * OPSZ
                        + lane * 16 + 8 * (nt & 1)) = e;
        }
    }
}

// ---------------------------------------------------------------------------
// attn : flash attention, one wave per (b, h, 16-query tile); 64 keys/iter.
// Row sums via WMMA against an all-ones B matrix (no sum-shuffles).
// Epilogue emits relu(out) as the Wp-stage B operand (f16), via LDS transpose.
// ---------------------------------------------------------------------------
#define AW 8
__global__ __launch_bounds__(AW * 32) void attn_kernel(
    const _Float16* __restrict__ qpk, const _Float16* __restrict__ kpk,
    const _Float16* __restrict__ vpk, _Float16* __restrict__ opk)
{
    __shared__ _Float16 ldsP[AW][64 * 16];          // per-wave P patch [key][query]
    const int wave = threadIdx.x >> 5, lane = threadIdx.x & 31;
    const int gw = blockIdx.x * AW + wave;          // 3456 waves
    const int qt = gw % NT;
    const int bh = gw / NT;
    const int lh = lane >> 4, lanelo = lane & 15;

    const v16h qa = *(const v16h*)(qpk + ((size_t)bh * NT + qt) * OPSZ + lane * 16);
    const _Float16* kbase = kpk + (size_t)bh * NT * OPSZ + lane * 16;
    const _Float16* vbase = vpk + (size_t)bh * (NT / 2) * 2 * OPSZ + lane * 16;
    _Float16* myP = &ldsP[wave][0];

    v16h vones;
    #pragma unroll
    for (int i = 0; i < 16; ++i) vones[i] = (_Float16)1.0f;

    v8f acc0 = {}, acc1 = {};                       // O: d 0..15 / 16..31
    float mrow[8], lrow[8], scl[8];
    #pragma unroll
    for (int v = 0; v < 8; ++v) { mrow[v] = -1e30f; lrow[v] = 0.0f; }

    for (int it = 0; it < NPIX / 64; ++it) {        // 36 iterations
        v16h kb0 = *(const v16h*)(kbase + (size_t)(4 * it + 0) * OPSZ);
        v16h kb1 = *(const v16h*)(kbase + (size_t)(4 * it + 1) * OPSZ);
        v16h kb2 = *(const v16h*)(kbase + (size_t)(4 * it + 2) * OPSZ);
        v16h kb3 = *(const v16h*)(kbase + (size_t)(4 * it + 3) * OPSZ);
        v8f z = {};
        v8f s0 = WMMA(qa, kb0, z);
        v8f s1 = WMMA(qa, kb1, z);
        v8f s2 = WMMA(qa, kb2, z);
        v8f s3 = WMMA(qa, kb3, z);

        v8h p0h, p1h, p2h, p3h;
        #pragma unroll
        for (int v = 0; v < 8; ++v) {               // row M = v + 8*lh
            float r = fmaxf(fmaxf(s0[v], s1[v]), fmaxf(s2[v], s3[v]));
            #pragma unroll
            for (int off = 1; off < 16; off <<= 1)
                r = fmaxf(r, __shfl_xor(r, off, 32));
            const float mnew = fmaxf(mrow[v], r);
            const float sc   = __expf(mrow[v] - mnew);
            mrow[v] = mnew; scl[v] = sc;
            acc0[v] *= sc; acc1[v] *= sc;
            p0h[v] = (_Float16)__expf(s0[v] - mnew);
            p1h[v] = (_Float16)__expf(s1[v] - mnew);
            p2h[v] = (_Float16)__expf(s2[v] - mnew);
            p3h[v] = (_Float16)__expf(s3[v] - mnew);
        }
        // store P transposed [key 64][query 16]: 4 x 16B per lane
        *(v8h*)(myP + (0  + lanelo) * 16 + 8 * lh) = p0h;
        *(v8h*)(myP + (16 + lanelo) * 16 + 8 * lh) = p1h;
        *(v8h*)(myP + (32 + lanelo) * 16 + 8 * lh) = p2h;
        *(v8h*)(myP + (48 + lanelo) * 16 + 8 * lh) = p3h;
        // gather P as A operands (keys 0..31 / 32..63)
        v16h paLo, paHi;
        #pragma unroll
        for (int i = 0; i < 16; ++i) {
            const int k = kmap(i, lh);
            paLo[i] = myP[k * 16 + lanelo];
            paHi[i] = myP[(32 + k) * 16 + lanelo];
        }
        // row sums of P via WMMA against ones (replaces 32 shuffles)
        v8f ls = WMMA(paLo, vones, z);
        ls     = WMMA(paHi, vones, ls);
        #pragma unroll
        for (int v = 0; v < 8; ++v) lrow[v] = lrow[v] * scl[v] + ls[v];
        // O += P @ V^T
        v16h vA0 = *(const v16h*)(vbase + (size_t)(4 * it + 0) * OPSZ);
        v16h vA1 = *(const v16h*)(vbase + (size_t)(4 * it + 1) * OPSZ);
        v16h vB0 = *(const v16h*)(vbase + (size_t)(4 * it + 2) * OPSZ);
        v16h vB1 = *(const v16h*)(vbase + (size_t)(4 * it + 3) * OPSZ);
        acc0 = WMMA(paLo, vA0, acc0);
        acc1 = WMMA(paLo, vA1, acc1);
        acc0 = WMMA(paHi, vB0, acc0);
        acc1 = WMMA(paHi, vB1, acc1);
    }

    // Finalize: 1/l, relu, transpose through LDS, emit Wp-stage operand tile.
    v8h o0, o1;
    #pragma unroll
    for (int v = 0; v < 8; ++v) {
        const float inv = 1.0f / lrow[v];
        o0[v] = (_Float16)fmaxf(acc0[v] * inv, 0.0f);
        o1[v] = (_Float16)fmaxf(acc1[v] * inv, 0.0f);
    }
    *(v8h*)(myP + (lanelo)      * 16 + 8 * lh) = o0;   // [d 32][query 16]
    *(v8h*)(myP + (16 + lanelo) * 16 + 8 * lh) = o1;
    v16h ov;
    #pragma unroll
    for (int i = 0; i < 16; ++i) ov[i] = myP[kmap(i, lh) * 16 + lanelo];
    const int b = bh / NH, h = bh % NH;                // kstep h covers d=h*32..+31
    *(v16h*)(opk + (((size_t)b * NT + qt) * NH + h) * OPSZ + lane * 16) = ov;
}

// ---------------------------------------------------------------------------
// gemm_p : xres = x + BN(Wp @ relu(out)); emits Wo-stage operand (f16).
// One wave per (b, n-tile, half): 3 co-tiles x 6 ksteps.
// ---------------------------------------------------------------------------
__global__ __launch_bounds__(256) void gemm_p_kernel(
    const _Float16* __restrict__ opk, const _Float16* __restrict__ wpp,
    const float* __restrict__ pbias, const float* __restrict__ xx,
    _Float16* __restrict__ xrpk)
{
    const int wave = threadIdx.x >> 5, lane = threadIdx.x & 31;
    const int gw = blockIdx.x * 8 + wave;              // 1152 waves
    const int g  = gw & 1;
    const int nt = (gw >> 1) % NT;
    const int b  = gw / (2 * NT);
    const int lh = lane >> 4, lanelo = lane & 15;

    v8f acc[3] = {{0}};
    for (int ks = 0; ks < NH; ++ks) {
        v16h bop = *(const v16h*)(opk + (((size_t)b * NT + nt) * NH + ks) * OPSZ + lane * 16);
        #pragma unroll
        for (int j = 0; j < 3; ++j) {
            const int t = g * 3 + j;
            v16h wa = *(const v16h*)(wpp + ((size_t)t * NH + ks) * OPSZ + lane * 16);
            acc[j] = WMMA(wa, bop, acc[j]);
        }
    }
    const int n = nt * 16 + lanelo;
    #pragma unroll
    for (int j = 0; j < 3; ++j) {
        const int t = g * 3 + j;
        const float* bp = pbias + t * 16 + 8 * lh;
        v8h e;
        #pragma unroll
        for (int v = 0; v < 8; ++v) {
            const int co = t * 16 + 8 * lh + v;
            const float res = xx[((size_t)b * CHAN + 3 * DIMS + co) * NPIX + n];
            e[v] = (_Float16)(acc[j][v] + bp[v] + res);
        }
        *(v8h*)(xrpk + (((size_t)b * NT + nt) * 3 + (t >> 1)) * OPSZ
                     + lane * 16 + 8 * (t & 1)) = e;
    }
}

// ---------------------------------------------------------------------------
// gemm_o : y = relu(BN(Wo @ xres)) -> (B, 384, 48, 48) fp32.
// One wave per (b, n-tile, half): 12 c-tiles x 3 ksteps.
// ---------------------------------------------------------------------------
__global__ __launch_bounds__(256) void gemm_o_kernel(
    const _Float16* __restrict__ xrpk, const _Float16* __restrict__ wop,
    const float* __restrict__ obias, float* __restrict__ y)
{
    const int wave = threadIdx.x >> 5, lane = threadIdx.x & 31;
    const int gw = blockIdx.x * 8 + wave;              // 1152 waves
    const int g  = gw & 1;
    const int nt = (gw >> 1) % NT;
    const int b  = gw / (2 * NT);
    const int lh = lane >> 4, lanelo = lane & 15;

    v8f acc[12] = {{0}};
    for (int ks = 0; ks < 3; ++ks) {
        v16h bop = *(const v16h*)(xrpk + (((size_t)b * NT + nt) * 3 + ks) * OPSZ + lane * 16);
        #pragma unroll
        for (int t = 0; t < 12; ++t) {
            const int ct = g * 12 + t;
            v16h wa = *(const v16h*)(wop + ((size_t)ct * 3 + ks) * OPSZ + lane * 16);
            acc[t] = WMMA(wa, bop, acc[t]);
        }
    }
    const int n = nt * 16 + lanelo;
    #pragma unroll
    for (int t = 0; t < 12; ++t) {
        const int ct = g * 12 + t;
        const float* bp = obias + ct * 16 + 8 * lh;
        #pragma unroll
        for (int v = 0; v < 8; ++v) {
            const int c = ct * 16 + 8 * lh + v;
            y[((size_t)b * CHAN + c) * NPIX + n] = fmaxf(acc[t][v] + bp[v], 0.0f);
        }
    }
}

// ---------------------------------------------------------------------------
// Launch
// ---------------------------------------------------------------------------
extern "C" void kernel_launch(void* const* d_in, const int* in_sizes, int n_in,
                              void* d_out, int out_size, void* d_ws, size_t ws_size,
                              hipStream_t stream)
{
    const float* xx = (const float*)d_in[0];
    const float* Wq = (const float*)d_in[1];
    const float* qs = (const float*)d_in[2];
    const float* qb = (const float*)d_in[3];
    const float* Wk = (const float*)d_in[4];
    const float* ks = (const float*)d_in[5];
    const float* kb = (const float*)d_in[6];
    const float* Wv = (const float*)d_in[7];
    const float* vs = (const float*)d_in[8];
    const float* vb = (const float*)d_in[9];
    const float* Wp = (const float*)d_in[10];
    const float* ps = (const float*)d_in[11];
    const float* pb = (const float*)d_in[12];
    const float* Wo = (const float*)d_in[13];
    const float* os = (const float*)d_in[14];
    const float* ob = (const float*)d_in[15];
    float* y = (float*)d_out;

    // workspace carve-up (f16 elements), ~23.4 MB total
    _Float16* p = (_Float16*)d_ws;
    const size_t xpk_n  = (size_t)BATCH * NT * XKS * OPSZ;        // 3,538,944
    const size_t wkp_n  = (size_t)6  * XKS * OPSZ;
    const size_t wqp_n  = (size_t)6  * 3   * OPSZ;
    const size_t wvp_n  = (size_t)12 * XKS * OPSZ;
    const size_t wpp_n  = (size_t)6  * NH  * OPSZ;
    const size_t wop_n  = (size_t)24 * 3   * OPSZ;
    const size_t kpk_n  = (size_t)BATCH * NH * NT * OPSZ;         // 1,769,472
    const size_t vpk_n  = (size_t)BATCH * NH * (NT / 2) * 2 * OPSZ;
    const size_t opk_n  = (size_t)BATCH * NT * NH * OPSZ;
    const size_t xrpk_n = (size_t)BATCH * NT * 3 * OPSZ;
    _Float16* xpk  = p; p += xpk_n;
    _Float16* wkp  = p; p += wkp_n;
    _Float16* wqp  = p; p += wqp_n;
    _Float16* wvp  = p; p += wvp_n;
    _Float16* wpp  = p; p += wpp_n;
    _Float16* wop  = p; p += wop_n;
    _Float16* kpk  = p; p += kpk_n;
    _Float16* qpk  = p; p += kpk_n;
    _Float16* vpk  = p; p += vpk_n;
    _Float16* opk  = p; p += opk_n;
    _Float16* xrpk = p;

    const int TPB = 256;
    pack_x_kernel<<<dim3((unsigned)(xpk_n / TPB)), TPB, 0, stream>>>(xx, xpk);
    pack_w_kernel<<<dim3((unsigned)(wkp_n / TPB)), TPB, 0, stream>>>(Wk, ks, wkp, 6,  XKS, CHAN);
    pack_w_kernel<<<dim3((unsigned)(wqp_n / TPB)), TPB, 0, stream>>>(Wq, qs, wqp, 6,  3,   DIMS);
    pack_w_kernel<<<dim3((unsigned)(wvp_n / TPB)), TPB, 0, stream>>>(Wv, vs, wvp, 12, XKS, CHAN);
    pack_w_kernel<<<dim3((unsigned)(wpp_n / TPB)), TPB, 0, stream>>>(Wp, ps, wpp, 6,  NH,  192);
    pack_w_kernel<<<dim3((unsigned)(wop_n / TPB)), TPB, 0, stream>>>(Wo, os, wop, 24, 3,   DIMS);

    gemm_qkv_kernel<<<dim3((BATCH * NT * 3) / 8), TPB, 0, stream>>>(
        xpk, wkp, wqp, wvp, kb, qb, vb, kpk, qpk, vpk);

    attn_kernel<<<dim3((BATCH * NH * NT) / AW), AW * 32, 0, stream>>>(qpk, kpk, vpk, opk);

    gemm_p_kernel<<<dim3((BATCH * NT * 2) / 8), TPB, 0, stream>>>(opk, wpp, pb, xx, xrpk);
    gemm_o_kernel<<<dim3((BATCH * NT * 2) / 8), TPB, 0, stream>>>(xrpk, wop, ob, y);
}
